// Block_51135880626810
// MI455X (gfx1250) — matrix operand order repaired and checked by reading, not compiled, not run on hardware
//
// CDNA5 (gfx1250) transformer block: LN -> QKV (WMMA bf16) -> flash attention
// (WMMA bf16, online softmax) -> proj+residual -> LN -> MLP (WMMA, LeakyReLU)
// -> residual. Compute-bound (~275 GFLOP vs ~0.2 GB unique traffic): all GEMMs
// run on v_wmma_f32_16x16x32_bf16 (f32 accum), tiles are staged with gfx1250
// GLOBAL_LOAD_ASYNC_TO_LDS and double-buffered so ASYNCcnt copies for tile t+1
// overlap the WMMA stream of tile t.

#include <hip/hip_runtime.h>

typedef __bf16 bf16;
typedef __attribute__((ext_vector_type(16))) __bf16 v16bf;
typedef __attribute__((ext_vector_type(8)))  __bf16 v8bf;
typedef __attribute__((ext_vector_type(8)))  float  v8f;

constexpr int Bb = 4, Nn = 2048, Cc = 1024, Hh = 4096, NHEADS = 16, HD = 64;
constexpr int Mrows = Bb * Nn;          // 8192 token rows
constexpr float LN_EPS = 1e-5f;

static __device__ __forceinline__ v8f wmma_bf16(v16bf a, v16bf b, v8f c) {
  return __builtin_amdgcn_wmma_f32_16x16x32_bf16(
      /*neg_a=*/false, a, /*neg_b=*/false, b,
      /*c_mod=*/(short)0, c, /*reuse_a=*/false, /*reuse_b=*/false);
}

// Build a v16bf fragment from two 16B-aligned LDS runs of 8 bf16 each.
// ISA 16-bit A 16x32 layout: lane L row = L%16; elems 0..7 -> K = koff+e,
// elems 8..15 -> K = koff+16+(e-8), koff = (L<16)?0:8.
// ISA 16-bit B 32x16 layout (transposed-in-LDS): lane col = L%16, same K
// mapping as A (kbase = (L<16)?0:8).  Both become two contiguous v8bf loads.
static __device__ __forceinline__ v16bf frag2(const bf16* lo, const bf16* hi) {
  v8bf l = *(const v8bf*)lo;
  v8bf h = *(const v8bf*)hi;
  v16bf r;
#pragma unroll
  for (int i = 0; i < 8; ++i) { r[i] = l[i]; r[i + 8] = h[i]; }
  return r;
}

// gfx1250 async copy: 16B global -> LDS, no VGPR data path, ASYNCcnt-tracked.
// GVS mode: mem addr = SGPR base + 32-bit VGPR byte offset (all offsets here
// are < 2^31). vdst VGPR carries the LDS byte address.
static __device__ __forceinline__ void async_cp16(void* lds_dst,
                                                  const bf16* gbase,
                                                  size_t elem_off) {
  unsigned lds = (unsigned)(uintptr_t)lds_dst;     // low 32 bits == LDS offset
  int goff = (int)(elem_off * sizeof(bf16));
  asm volatile("global_load_async_to_lds_b128 %0, %1, %2"
               :: "v"(lds), "v"(goff), "s"(gbase)
               : "memory");
}
static __device__ __forceinline__ void wait_async() {
  asm volatile("s_wait_asynccnt 0x0" ::: "memory");
}

union U4 { uint2 u; bf16 e[4]; };

// ---------------------------------------------------------------- fp32->bf16
__global__ void cvt_bf16_kernel(const float* __restrict__ s,
                                bf16* __restrict__ d, int n) {
  int i = blockIdx.x * blockDim.x + threadIdx.x;
  int stride = gridDim.x * blockDim.x;
  for (; i < n; i += stride) d[i] = (bf16)s[i];
}

// ------------------------------------------------- triple LayerNorm (QKV in)
__global__ __launch_bounds__(256) void ln3_kernel(
    const float* __restrict__ x, const float* __restrict__ pos,
    const float* __restrict__ gq, const float* __restrict__ bq,
    const float* __restrict__ gk, const float* __restrict__ bk,
    const float* __restrict__ gv, const float* __restrict__ bv,
    bf16* __restrict__ lnq, bf16* __restrict__ lnk, bf16* __restrict__ lnv) {
  int row = blockIdx.x;
  int n = row % Nn;
  const float* xr = x + (size_t)row * Cc;
  const float* pr = pos + (size_t)n * Cc;
  int t = threadIdx.x, lane = t & 31, wid = t >> 5;

  float v0[4], v1[4];
  float s0 = 0.f, q0 = 0.f, s1 = 0.f, q1 = 0.f;
#pragma unroll
  for (int i = 0; i < 4; ++i) {
    int c = t + i * 256;
    float a = xr[c], ap = a + pr[c];
    v0[i] = a; v1[i] = ap;
    s0 += a; q0 += a * a; s1 += ap; q1 += ap * ap;
  }
#pragma unroll
  for (int m = 16; m >= 1; m >>= 1) {
    s0 += __shfl_xor(s0, m, 32); q0 += __shfl_xor(q0, m, 32);
    s1 += __shfl_xor(s1, m, 32); q1 += __shfl_xor(q1, m, 32);
  }
  __shared__ float red[4][8];
  if (lane == 0) { red[0][wid] = s0; red[1][wid] = q0; red[2][wid] = s1; red[3][wid] = q1; }
  __syncthreads();
  float S0 = 0.f, Q0 = 0.f, S1 = 0.f, Q1 = 0.f;
#pragma unroll
  for (int w = 0; w < 8; ++w) { S0 += red[0][w]; Q0 += red[1][w]; S1 += red[2][w]; Q1 += red[3][w]; }
  float m0 = S0 / Cc, m1 = S1 / Cc;
  float r0 = rsqrtf(Q0 / Cc - m0 * m0 + LN_EPS);
  float r1 = rsqrtf(Q1 / Cc - m1 * m1 + LN_EPS);
#pragma unroll
  for (int i = 0; i < 4; ++i) {
    int c = t + i * 256;
    size_t o = (size_t)row * Cc + c;
    lnq[o] = (bf16)((v0[i] - m0) * r0 * gq[c] + bq[c]);
    float kn = (v1[i] - m1) * r1;
    lnk[o] = (bf16)(kn * gk[c] + bk[c]);
    lnv[o] = (bf16)(kn * gv[c] + bv[c]);
  }
}

// --------------------------------------------------------- single LayerNorm
__global__ __launch_bounds__(256) void ln1_kernel(
    const float* __restrict__ in, const float* __restrict__ g,
    const float* __restrict__ b, bf16* __restrict__ out) {
  int row = blockIdx.x;
  const float* xr = in + (size_t)row * Cc;
  int t = threadIdx.x, lane = t & 31, wid = t >> 5;
  float v0[4], s0 = 0.f, q0 = 0.f;
#pragma unroll
  for (int i = 0; i < 4; ++i) {
    float a = xr[t + i * 256]; v0[i] = a; s0 += a; q0 += a * a;
  }
#pragma unroll
  for (int m = 16; m >= 1; m >>= 1) {
    s0 += __shfl_xor(s0, m, 32); q0 += __shfl_xor(q0, m, 32);
  }
  __shared__ float red[2][8];
  if (lane == 0) { red[0][wid] = s0; red[1][wid] = q0; }
  __syncthreads();
  float S0 = 0.f, Q0 = 0.f;
#pragma unroll
  for (int w = 0; w < 8; ++w) { S0 += red[0][w]; Q0 += red[1][w]; }
  float m0 = S0 / Cc;
  float r0 = rsqrtf(Q0 / Cc - m0 * m0 + LN_EPS);
#pragma unroll
  for (int i = 0; i < 4; ++i) {
    int c = t + i * 256;
    out[(size_t)row * Cc + c] = (bf16)((v0[i] - m0) * r0 * g[c] + b[c]);
  }
}

// --------------------------------------------------------------- WMMA GEMM
// C[M x Nout] = A[M x K](bf16) @ W[K x Nout](bf16) + bias, optional LeakyReLU,
// optional fp32 residual add. Block = 128 thr (4 waves), tile 128x64, BK=32.
// Each wave owns a 32x64 strip: 2 A-frags share 4 B-frags -> 8 WMMA / k-step.
// Double-buffered: async A copies + B global loads for tile t+1 are issued
// before the WMMAs of tile t, drained (s_wait_asynccnt) only at iteration end.
template <int ACT, bool RESID, bool OUTBF>
__global__ __launch_bounds__(128) void gemm_kernel(
    const bf16* __restrict__ A, const bf16* __restrict__ W,
    const float* __restrict__ bias, const float* __restrict__ resid,
    float* __restrict__ outf, bf16* __restrict__ outb,
    int Md, int K, int Nout) {
  constexpr int BM = 128, BN = 64, BK = 32;
  __shared__ __align__(16) bf16 As[2][BM][BK + 8];  // [m][k], stride 40 (80B)
  __shared__ __align__(16) bf16 Bs[2][BN][BK + 8];  // transposed: [n][k]
  int bm = blockIdx.x * BM, bn = blockIdx.y * BN;
  int tid = threadIdx.x, lane = tid & 31, wave = tid >> 5;

  v8f acc0[4] = {}, acc1[4] = {};
  const int ar0 = wave * 32 + (lane & 15);
  const int ar1 = ar0 + 16;
  const int koff = (lane < 16) ? 0 : 8;
  const int kbase = (lane < 16) ? 0 : 8;
  const int nk = K / BK;

  U4 breg[4];
  auto issueA = [&](int kt, int buf) {
#pragma unroll
    for (int p = 0; p < 4; ++p) {
      int idx = tid + p * 128;
      int r = idx >> 2;            // 0..127
      int c = (idx & 3) * 8;       // 0,8,16,24
      async_cp16(&As[buf][r][c], A, (size_t)(bm + r) * K + kt * BK + c);
    }
  };
  auto loadB = [&](int kt) {
#pragma unroll
    for (int p = 0; p < 4; ++p) {
      int kr = (tid >> 4) + p * 8;
      int c = (tid & 15) * 4;
      breg[p].u = *(const uint2*)(W + (size_t)(kt * BK + kr) * Nout + bn + c);
    }
  };
  auto storeB = [&](int buf) {
#pragma unroll
    for (int p = 0; p < 4; ++p) {
      int kr = (tid >> 4) + p * 8;
      int c = (tid & 15) * 4;
#pragma unroll
      for (int i = 0; i < 4; ++i) Bs[buf][c + i][kr] = breg[p].e[i];
    }
  };

  // Prologue: stage tile 0.
  issueA(0, 0);
  loadB(0);
  storeB(0);
  wait_async();
  __syncthreads();

  for (int kt = 0; kt < nk; ++kt) {
    int cur = kt & 1, nxt = cur ^ 1;
    bool more = (kt + 1 < nk);
    if (more) { issueA(kt + 1, nxt); loadB(kt + 1); }   // overlap with WMMAs

    v16bf a0 = frag2(&As[cur][ar0][koff], &As[cur][ar0][koff + 16]);
    v16bf a1 = frag2(&As[cur][ar1][koff], &As[cur][ar1][koff + 16]);
#pragma unroll
    for (int nt = 0; nt < 4; ++nt) {
      int bc = nt * 16 + (lane & 15);
      v16bf b = frag2(&Bs[cur][bc][kbase], &Bs[cur][bc][kbase + 16]);
      acc0[nt] = wmma_bf16(a0, b, acc0[nt]);
      acc1[nt] = wmma_bf16(a1, b, acc1[nt]);
    }

    if (more) storeB(nxt);
    wait_async();        // next-tile async copies done (issued ~8 WMMAs ago)
    __syncthreads();
  }

  // C/D layout: VGPR r, lane L -> row = r + 8*(L>=16), col = L%16.
  const int rh = (lane >> 4) * 8;
#pragma unroll
  for (int nt = 0; nt < 4; ++nt) {
    int col = bn + nt * 16 + (lane & 15);
    float bv = bias[col];
#pragma unroll
    for (int r = 0; r < 8; ++r) {
#pragma unroll
      for (int sub = 0; sub < 2; ++sub) {
        int grow = bm + wave * 32 + sub * 16 + rh + r;
        float val = (sub ? acc1[nt][r] : acc0[nt][r]) + bv;
        if (ACT == 1) val = (val >= 0.f) ? val : 0.1f * val;  // LeakyReLU(0.1)
        size_t idx = (size_t)grow * Nout + col;
        if constexpr (RESID) outf[idx] = resid[idx] + val;
        if constexpr (OUTBF) outb[idx] = (bf16)val;
      }
    }
  }
}

// ------------------------------------------------------- flash attention
// grid = (B*HEADS, N/64), block = 128 thr (4 waves). Each block: one head,
// 64 query rows; streams 32 key blocks of 64 with online softmax in LDS.
// K tiles async-to-LDS, V tiles transposed via VGPRs; both double-buffered so
// tile t+1 traffic overlaps the S=QK^T / softmax / O+=PV work of tile t.
__global__ __launch_bounds__(128) void attn_kernel(
    const bf16* __restrict__ Qg, const bf16* __restrict__ Kg,
    const bf16* __restrict__ Vg, bf16* __restrict__ Og) {
  __shared__ __align__(16) bf16 Qs[64][72];       // [q][d]
  __shared__ __align__(16) bf16 Ks[2][64][72];    // [key][d] (== B^T layout)
  __shared__ __align__(16) bf16 Vs[2][64][72];    // transposed: [d][key]
  __shared__ __align__(16) bf16 Ss[64][72];       // P bf16 [q][key], wave-private rows

  int bh = blockIdx.x;
  int b = bh / NHEADS, h = bh % NHEADS;
  int qt = blockIdx.y;
  int tid = threadIdx.x, lane = tid & 31, wave = tid >> 5;
  size_t qrow0 = (size_t)b * Nn + qt * 64;
  int cb = h * HD;

  U4 vreg[8];
  auto issueK = [&](int kb, int buf) {
#pragma unroll
    for (int p = 0; p < 4; ++p) {
      int idx = tid + p * 128;
      int r = idx >> 3;            // 0..63
      int c = (idx & 7) * 8;       // 0..56
      async_cp16(&Ks[buf][r][c], Kg, ((size_t)b * Nn + kb + r) * Cc + cb + c);
    }
  };
  auto loadV = [&](int kb) {
#pragma unroll
    for (int p = 0; p < 8; ++p) {
      int r = (tid >> 4) + p * 8;
      int c = (tid & 15) * 4;
      vreg[p].u = *(const uint2*)(Vg + ((size_t)b * Nn + kb + r) * Cc + cb + c);
    }
  };
  auto storeV = [&](int buf) {
#pragma unroll
    for (int p = 0; p < 8; ++p) {
      int r = (tid >> 4) + p * 8;
      int c = (tid & 15) * 4;
#pragma unroll
      for (int i = 0; i < 4; ++i) Vs[buf][c + i][r] = vreg[p].e[i];
    }
  };

  // Q tile 64x64 async (stays resident the whole kernel).
#pragma unroll
  for (int p = 0; p < 4; ++p) {
    int idx = tid + p * 128;
    int r = idx >> 3;
    int c = (idx & 7) * 8;
    async_cp16(&Qs[r][c], Qg, (qrow0 + r) * Cc + cb + c);
  }
  // Prologue: stage key/value tile 0.
  issueK(0, 0);
  loadV(0);
  storeV(0);
  wait_async();
  __syncthreads();

  const int arow = wave * 16 + (lane & 15);
  const int koff = (lane < 16) ? 0 : 8;
  const int kbase = (lane < 16) ? 0 : 8;
  const int rh = (lane >> 4) * 8;
  const float scale = 0.125f;  // 1/sqrt(64)

  float Mr[8], Lr[8];
  v8f Oacc[4] = {};
#pragma unroll
  for (int r = 0; r < 8; ++r) { Mr[r] = -INFINITY; Lr[r] = 0.f; }

  constexpr int NKB = Nn / 64;
  for (int t = 0; t < NKB; ++t) {
    int cur = t & 1, nxt = cur ^ 1;
    bool more = (t + 1 < NKB);
    if (more) { issueK((t + 1) * 64, nxt); loadV((t + 1) * 64); }

    // S = Q @ K^T  (K-dim = hd = 64 -> two 32-deep WMMA steps)
    v8f Sacc[4] = {};
#pragma unroll
    for (int ks = 0; ks < 64; ks += 32) {
      v16bf a = frag2(&Qs[arow][ks + koff], &Qs[arow][ks + koff + 16]);
#pragma unroll
      for (int nt = 0; nt < 4; ++nt) {
        int key = nt * 16 + (lane & 15);
        v16bf bm = frag2(&Ks[cur][key][ks + kbase], &Ks[cur][key][ks + kbase + 16]);
        Sacc[nt] = wmma_bf16(a, bm, Sacc[nt]);
      }
    }

    // Online softmax. Each row's 16 cols/tile live in one half-wave, so
    // shfl_xor masks 1,2,4,8 reduce across the row.
#pragma unroll
    for (int r = 0; r < 8; ++r) {
      float mx = -INFINITY;
#pragma unroll
      for (int nt = 0; nt < 4; ++nt) mx = fmaxf(mx, Sacc[nt][r]);
      mx *= scale;
#pragma unroll
      for (int m = 1; m <= 8; m <<= 1) mx = fmaxf(mx, __shfl_xor(mx, m, 32));
      float newm = fmaxf(Mr[r], mx);
      float alpha = __expf(Mr[r] - newm);
      float rsum = 0.f;
      int mrow = wave * 16 + rh + r;
#pragma unroll
      for (int nt = 0; nt < 4; ++nt) {
        float p = __expf(Sacc[nt][r] * scale - newm);
        rsum += p;
        Ss[mrow][nt * 16 + (lane & 15)] = (bf16)p;
        Oacc[nt][r] *= alpha;
      }
#pragma unroll
      for (int m = 1; m <= 8; m <<= 1) rsum += __shfl_xor(rsum, m, 32);
      Lr[r] = Lr[r] * alpha + rsum;
      Mr[r] = newm;
    }
    // Each wave reads back only its own 16 rows of Ss -> wave-local LDS RAW.
    asm volatile("s_wait_dscnt 0" ::: "memory");

    // O += P @ V  (K-dim = 64 keys)
#pragma unroll
    for (int ks = 0; ks < 64; ks += 32) {
      v16bf a = frag2(&Ss[arow][ks + koff], &Ss[arow][ks + koff + 16]);
#pragma unroll
      for (int nt = 0; nt < 4; ++nt) {
        int dcol = nt * 16 + (lane & 15);
        v16bf bm = frag2(&Vs[cur][dcol][ks + kbase], &Vs[cur][dcol][ks + kbase + 16]);
        Oacc[nt] = wmma_bf16(a, bm, Oacc[nt]);
      }
    }

    if (more) storeV(nxt);
    wait_async();        // next K tile copies done (issued 16 WMMAs ago)
    __syncthreads();
  }

  // Normalize and write out bf16.
#pragma unroll
  for (int nt = 0; nt < 4; ++nt) {
#pragma unroll
    for (int r = 0; r < 8; ++r) {
      int mrow = wave * 16 + rh + r;
      float val = Oacc[nt][r] / Lr[r];
      Og[(qrow0 + mrow) * Cc + cb + nt * 16 + (lane & 15)] = (bf16)val;
    }
  }
}

// ---------------------------------------------------------------- launcher
extern "C" void kernel_launch(void* const* d_in, const int* in_sizes, int n_in,
                              void* d_out, int out_size, void* d_ws,
                              size_t ws_size, hipStream_t stream) {
  const float* x     = (const float*)d_in[0];
  const float* pos   = (const float*)d_in[1];
  const float* nq_g  = (const float*)d_in[2];
  const float* nq_b  = (const float*)d_in[3];
  const float* nk_g  = (const float*)d_in[4];
  const float* nk_b  = (const float*)d_in[5];
  const float* nv_g  = (const float*)d_in[6];
  const float* nv_b  = (const float*)d_in[7];
  const float* wq    = (const float*)d_in[8];
  const float* bq    = (const float*)d_in[9];
  const float* wk    = (const float*)d_in[10];
  const float* bk    = (const float*)d_in[11];
  const float* wv    = (const float*)d_in[12];
  const float* bv    = (const float*)d_in[13];
  const float* wp    = (const float*)d_in[14];
  const float* bp    = (const float*)d_in[15];
  const float* n_g   = (const float*)d_in[16];
  const float* n_b   = (const float*)d_in[17];
  const float* w1    = (const float*)d_in[18];
  const float* b1    = (const float*)d_in[19];
  const float* w2    = (const float*)d_in[20];
  const float* b2    = (const float*)d_in[21];
  float* out = (float*)d_out;

  // Workspace carve-up (all offsets 256B aligned).
  char* ws = (char*)d_ws;
  size_t off = 0;
  auto take = [&](size_t bytes) {
    size_t r = off;
    off += (bytes + 255) & ~(size_t)255;
    return r;
  };
  const size_t MC = (size_t)Mrows * Cc, MH = (size_t)Mrows * Hh;
  bf16* wq_b = (bf16*)(ws + take((size_t)Cc * Cc * 2));
  bf16* wk_b = (bf16*)(ws + take((size_t)Cc * Cc * 2));
  bf16* wv_b = (bf16*)(ws + take((size_t)Cc * Cc * 2));
  bf16* wp_b = (bf16*)(ws + take((size_t)Cc * Cc * 2));
  bf16* w1_b = (bf16*)(ws + take((size_t)Cc * Hh * 2));
  bf16* w2_b = (bf16*)(ws + take((size_t)Hh * Cc * 2));
  bf16* lnq  = (bf16*)(ws + take(MC * 2));
  bf16* lnk  = (bf16*)(ws + take(MC * 2));
  bf16* lnv  = (bf16*)(ws + take(MC * 2));
  bf16* qb   = (bf16*)(ws + take(MC * 2));
  bf16* kbuf = (bf16*)(ws + take(MC * 2));
  bf16* vbuf = (bf16*)(ws + take(MC * 2));
  bf16* ab   = (bf16*)(ws + take(MC * 2));   // attention output
  float* hbuf = (float*)(ws + take(MC * 4)); // residual h
  bf16* lnh  = (bf16*)(ws + take(MC * 2));
  bf16* y1   = (bf16*)(ws + take(MH * 2));   // MLP hidden

  // 1) weights fp32 -> bf16 (once per call; 24 MB, lives in 192 MB L2)
  cvt_bf16_kernel<<<2048, 256, 0, stream>>>(wq, wq_b, Cc * Cc);
  cvt_bf16_kernel<<<2048, 256, 0, stream>>>(wk, wk_b, Cc * Cc);
  cvt_bf16_kernel<<<2048, 256, 0, stream>>>(wv, wv_b, Cc * Cc);
  cvt_bf16_kernel<<<2048, 256, 0, stream>>>(wp, wp_b, Cc * Cc);
  cvt_bf16_kernel<<<4096, 256, 0, stream>>>(w1, w1_b, Cc * Hh);
  cvt_bf16_kernel<<<4096, 256, 0, stream>>>(w2, w2_b, Hh * Cc);

  // 2) fused triple LayerNorm
  ln3_kernel<<<Mrows, 256, 0, stream>>>(x, pos, nq_g, nq_b, nk_g, nk_b,
                                        nv_g, nv_b, lnq, lnk, lnv);

  // 3) Q/K/V projections (WMMA)
  dim3 gqkv(Mrows / 128, Cc / 64);
  gemm_kernel<0, false, true><<<gqkv, 128, 0, stream>>>(
      lnq, wq_b, bq, nullptr, nullptr, qb, Mrows, Cc, Cc);
  gemm_kernel<0, false, true><<<gqkv, 128, 0, stream>>>(
      lnk, wk_b, bk, nullptr, nullptr, kbuf, Mrows, Cc, Cc);
  gemm_kernel<0, false, true><<<gqkv, 128, 0, stream>>>(
      lnv, wv_b, bv, nullptr, nullptr, vbuf, Mrows, Cc, Cc);

  // 4) flash attention
  attn_kernel<<<dim3(Bb * NHEADS, Nn / 64), 128, 0, stream>>>(qb, kbuf, vbuf, ab);

  // 5) output projection + residual: h = x + ab @ wp + bp  (fp32)
  gemm_kernel<0, true, false><<<gqkv, 128, 0, stream>>>(
      ab, wp_b, bp, x, hbuf, nullptr, Mrows, Cc, Cc);

  // 6) LN(h)
  ln1_kernel<<<Mrows, 256, 0, stream>>>(hbuf, n_g, n_b, lnh);

  // 7) MLP up + LeakyReLU(0.1)
  gemm_kernel<1, false, true><<<dim3(Mrows / 128, Hh / 64), 128, 0, stream>>>(
      lnh, w1_b, b1, nullptr, nullptr, y1, Mrows, Cc, Hh);

  // 8) MLP down + residual -> final output (fp32)
  gemm_kernel<0, true, false><<<gqkv, 128, 0, stream>>>(
      y1, w2_b, b2, hbuf, out, nullptr, Mrows, Hh, Cc);
}
